// RTN_79173427134905
// MI455X (gfx1250) — matrix operand order repaired and checked by reading, not compiled, not run on hardware
//
#include <hip/hip_runtime.h>

typedef __bf16 bf16_t;
typedef __attribute__((ext_vector_type(16))) __bf16 v16bf;
typedef __attribute__((ext_vector_type(8)))  __bf16 v8bf;
typedef __attribute__((ext_vector_type(8)))  float  v8f;

#define BATCH   1024
#define WSZ     19
#define PPAIR   171
#define HD      256
#define GD      256
#define NROWS   (BATCH*PPAIR)          // 175104, divisible by 16
#define EPSV    1e-6f

// ---- d_out layout (flat concat in reference return order) ----
#define H_OFF     0
#define H_SIZE    (BATCH*WSZ*128)
#define KLB_OFF   (H_OFF + H_SIZE)
#define KLG_OFF   (KLB_OFF + 1)
#define SKLB_OFF  (KLG_OFF + 1)
#define SKL_SIZE  (NROWS*GD)
#define SKLG_OFF  (SKLB_OFF + SKL_SIZE)
#define SUMM_OFF  (SKLG_OFF + SKL_SIZE)
#define ALPHA_OFF (SUMM_OFF + NROWS)

// ---- LDS row strides (elements), padded for bank-conflict-free frag loads,
//      16-byte aligned rows for ds_load_b128 ----
#define SSTR 520   // 512 + 8 bf16
#define ASTR 264   // 256 + 8 bf16
#define FSTR 260   // 256 + 4 f32

__device__ __forceinline__ float softplus_f(float v) {
  return v > 20.f ? v : log1pf(__expf(v));
}

// ---------------- WMMA 16xN GEMM layer: dst = act(src[16xK] @ Wt^T + bias) ----------------
// src: LDS bf16 row-major [16][K] (stride sstr). wt: global bf16, transposed [256][K].
// 8 waves; wave w computes output cols [32w, 32w+32) as two 16x16 WMMA tiles.
// Single k-loop: one A-frag load feeds BOTH N-tile accumulators (halves LDS traffic,
// keeps two independent WMMA chains in flight for latency hiding).
template<int KDIM, int ACT, int DSTF32>
__device__ __forceinline__ void gemm16(const bf16_t* __restrict__ src, int sstr,
                                       const bf16_t* __restrict__ wt,
                                       const float*  __restrict__ bias,
                                       bf16_t* dstb, float* dstf, int dstr, int tid)
{
  const int lane = tid & 31;
  const int wave = tid >> 5;
  const int row  = lane & 15;   // A: M row / B: N col / C: N col owned by this lane
  const int hi   = lane >> 4;
  const bf16_t* arow  = src + row * sstr + hi * 16;
  const bf16_t* wcol0 = wt + (size_t)(wave * 32 + row) * KDIM + hi * 16;
  const bf16_t* wcol1 = wcol0 + (size_t)16 * KDIM;
  v8f acc0 = 0.0f, acc1 = 0.0f;
#pragma unroll
  for (int kb = 0; kb < KDIM; kb += 32) {
    v16bf av;
    // A 16x32 layout (matches B): elems 0..15 -> K = kb+hi*16+0..15
    *((v8bf*)&av)       = *(const v8bf*)(arow + kb);
    *(((v8bf*)&av) + 1) = *(const v8bf*)(arow + kb + 8);
    // B 32x16: lane holds col (lane&15), K = hi*16 + e, contiguous in transposed W
    v16bf b0 = *(const v16bf*)(wcol0 + kb);
    v16bf b1 = *(const v16bf*)(wcol1 + kb);
    acc0 = __builtin_amdgcn_wmma_f32_16x16x32_bf16(false, av, false, b0,
                                                   (short)0, acc0, false, false);
    acc1 = __builtin_amdgcn_wmma_f32_16x16x32_bf16(false, av, false, b1,
                                                   (short)0, acc1, false, false);
  }
#pragma unroll
  for (int t = 0; t < 2; ++t) {
    const v8f&  acc = t ? acc1 : acc0;
    const int   n   = wave * 32 + t * 16 + row;
    const float bb  = bias[n];
#pragma unroll
    for (int r = 0; r < 8; ++r) {
      const int m = r + hi * 8;               // C layout: VGPR r, lane-hi selects M block
      float v = acc[r] + bb;
      if constexpr (ACT == 1) v = fmaxf(v, 0.f);
      if constexpr (ACT == 2) v = softplus_f(v);
      if constexpr (DSTF32) dstf[m * dstr + n] = v;
      else                  dstb[m * dstr + n] = (bf16_t)v;
    }
  }
}

// ---------------- 16-row GEMV (256 -> 1): out[r] = oact(dot(src[r], w) + b) ----------------
template<int OACT, typename ST>
__device__ __forceinline__ void gemv16(const ST* src, int sstr,
                                       const float* __restrict__ w,
                                       const float* __restrict__ bptr,
                                       float* outp, int ostr, float* sred, int tid)
{
  const int r = tid & 15, cb = (tid >> 4) * 16;
  float s = 0.f;
#pragma unroll
  for (int i = 0; i < 16; ++i) s += (float)src[r * sstr + cb + i] * w[cb + i];
  sred[tid] = s;                      // layout: [chunk*16 + r]
  __syncthreads();
  if (tid < 16) {
    float t = bptr[0];
    for (int i = 0; i < 16; ++i) t += sred[i * 16 + tid];
    if constexpr (OACT == 1) t = softplus_f(t);
    if constexpr (OACT == 2) t = 0.4f / (1.f + __expf(-t));
    if constexpr (OACT == 3) t = fmaxf(t, 0.f);
    outp[tid * ostr] = t;
  }
  __syncthreads();
}

struct P256 { const bf16_t* wt; const float* b; };
struct Scal { const float* w; const float* b; };
struct MegaArgs {
  const float* x; const float* eps1; float* out;
  P256 L[11];   // edge0,edge1,edge2, pr.enc,pr.enc_g,pr.enc_b, po.enc,po.enc_g,po.enc_b, pma,psa
  Scal pmean, pstd, pmij, qmean, qstd, gtrans;
};

__global__ void __launch_bounds__(256) mega_kernel(MegaArgs a)
{
  __shared__ __align__(16) bf16_t sIn [16 * SSTR];
  __shared__ __align__(16) bf16_t sA  [16 * ASTR];
  __shared__ __align__(16) bf16_t sB  [16 * ASTR];
  __shared__ __align__(16) bf16_t sEmb[16 * ASTR];
  __shared__ float sPma[16 * FSTR];
  __shared__ float sPsa[16 * FSTR];
  __shared__ float sMij[16 * FSTR];
  __shared__ float sRed[256];
  __shared__ float sScal[16 * 8];     // 0 pm, 1 ps, 2 pmij, 3 qm, 4 qs, 5 eps1
  __shared__ int   sOff[16 * 2];

  const int tid = threadIdx.x;
  const int g0  = blockIdx.x * 16;

  if (tid < 16) {
    int g = g0 + tid;
    int b = g / PPAIR, p = g - b * PPAIR;
    int j = 1; while (((j * (j + 1)) >> 1) <= p) ++j;   // p = j(j-1)/2 + k
    int k = p - ((j * (j - 1)) >> 1);
    sOff[tid * 2 + 0] = (b * WSZ + j) * HD;
    sOff[tid * 2 + 1] = (b * WSZ + k) * HD;
    sScal[tid * 8 + 5] = a.eps1[g];
  }
  __syncthreads();
  // node_pairs tile, f32 -> bf16; vectorized: 16*128 float4 loads
  for (int idx = tid; idx < 16 * 128; idx += 256) {
    int r = idx >> 7, c4 = (idx & 127) * 4;
    int off = (c4 < 256) ? (sOff[r * 2] + c4) : (sOff[r * 2 + 1] + c4 - 256);
    float4 v = *(const float4*)(a.x + off);
    bf16_t* dst = &sIn[r * SSTR + c4];
    dst[0] = (bf16_t)v.x; dst[1] = (bf16_t)v.y;
    dst[2] = (bf16_t)v.z; dst[3] = (bf16_t)v.w;
  }
  __syncthreads();

  gemm16<512,1,0>(sIn,  SSTR, a.L[0].wt,  a.L[0].b,  sA,  nullptr, ASTR, tid); __syncthreads();
  gemm16<256,1,0>(sA,   ASTR, a.L[1].wt,  a.L[1].b,  sB,  nullptr, ASTR, tid); __syncthreads();
  gemm16<256,0,0>(sB,   ASTR, a.L[2].wt,  a.L[2].b,  sEmb,nullptr, ASTR, tid); __syncthreads();
  // prior branch
  gemm16<256,1,0>(sEmb, ASTR, a.L[3].wt,  a.L[3].b,  sA,  nullptr, ASTR, tid); __syncthreads();
  gemm16<256,1,0>(sA,   ASTR, a.L[4].wt,  a.L[4].b,  sB,  nullptr, ASTR, tid); __syncthreads();
  gemv16<0>(sB, ASTR, a.pmean.w, a.pmean.b, &sScal[0], 8, sRed, tid);
  gemv16<1>(sB, ASTR, a.pstd.w,  a.pstd.b,  &sScal[1], 8, sRed, tid);
  gemm16<256,1,0>(sA,   ASTR, a.L[5].wt,  a.L[5].b,  sB,  nullptr, ASTR, tid); __syncthreads();
  gemv16<2>(sB, ASTR, a.pmij.w,  a.pmij.b,  &sScal[2], 8, sRed, tid);
  // posterior branch
  gemm16<256,1,0>(sEmb, ASTR, a.L[6].wt,  a.L[6].b,  sA,  nullptr, ASTR, tid); __syncthreads();
  gemm16<256,1,0>(sA,   ASTR, a.L[7].wt,  a.L[7].b,  sB,  nullptr, ASTR, tid); __syncthreads();
  gemv16<0>(sB, ASTR, a.qmean.w, a.qmean.b, &sScal[3], 8, sRed, tid);
  gemv16<1>(sB, ASTR, a.qstd.w,  a.qstd.b,  &sScal[4], 8, sRed, tid);
  gemm16<256,1,0>(sA,   ASTR, a.L[8].wt,  a.L[8].b,  sB,  nullptr, ASTR, tid); __syncthreads();
  gemm16<256,0,1>(sB,   ASTR, a.L[9].wt,  a.L[9].b,  nullptr, sPma, FSTR, tid); __syncthreads();
  gemm16<256,2,1>(sB,   ASTR, a.L[10].wt, a.L[10].b, nullptr, sPsa, FSTR, tid); __syncthreads();

  // ---- fused elementwise: post_mij, alpha_tilde, S_kl_g, S_kl_b, kl partials ----
  float klb = 0.f, klg = 0.f;
#pragma unroll 4
  for (int r = 0; r < 16; ++r) {
    const int c = tid;
    float pmaV = sPma[r * FSTR + c];
    float psaV = sPsa[r * FSTR + c];
    float nij  = 2.f * pmaV - 1.f;
    float mij  = 0.25f * (nij + __fsqrt_rn(nij * nij + 8.f * psaV * psaV)) + EPSV;
    sMij[r * FSTR + c] = mij;
    float e1 = sScal[r * 8 + 5];
    float at = softplus_f(e1 * __fsqrt_rn(mij) + mij);
    float pm = sScal[r * 8 + 0], ps = sScal[r * 8 + 1], pj = sScal[r * 8 + 2];
    float qm = sScal[r * 8 + 3], qs = sScal[r * 8 + 4];
    float sat = __fsqrt_rn(at);
    float a1 = at * qm, a2 = sat * qs, a3 = at * pm, a4 = sat * ps;
    float a4e = a4 + EPSV, d13 = a1 - a3;
    float kld = 2.f * __logf(a4e) - 2.f * __logf(a2 + EPSV)
              + (a2 * a2 + d13 * d13) / (a4e * a4e) - 1.f;
    float sklg = 0.5f * kld;
    float first  = mij * (__logf(mij + EPSV) - __logf(pj + EPSV));
    float second = (1.f - mij) * (__logf(1.f - mij + 0.5f * mij * mij + EPSV)
                                - __logf(1.f - pj  + 0.5f * pj  * pj  + EPSV));
    float sklb = first + second;
    klg += fabsf(sklg);
    klb += fabsf(sklb);
    const size_t g = (size_t)(g0 + r);
    a.out[SKLB_OFF + g * GD + c] = sklb;
    a.out[SKLG_OFF + g * GD + c] = sklg;
  }
  __syncthreads();

  // summ_graph = mean over 256 dims of post_mij
  {
    const int r = tid >> 4, cb = (tid & 15) * 16;
    float s = 0.f;
#pragma unroll
    for (int i = 0; i < 16; ++i) s += sMij[r * FSTR + cb + i];
    sRed[tid] = s;
    __syncthreads();
    if (tid < 16) {
      float t = 0.f;
      for (int i = 0; i < 16; ++i) t += sRed[tid * 16 + i];
      a.out[SUMM_OFF + g0 + tid] = t * (1.f / 256.f);
    }
    __syncthreads();
  }

  // alpha_bar = relu(graph_trans(post_mij))
  gemv16<3>(sMij, FSTR, a.gtrans.w, a.gtrans.b, a.out + ALPHA_OFF + g0, 1, sRed, tid);

  // block-reduce kl partials -> global atomics
  sRed[tid] = klb; __syncthreads();
  for (int s = 128; s > 0; s >>= 1) { if (tid < s) sRed[tid] += sRed[tid + s]; __syncthreads(); }
  if (tid == 0) atomicAdd(a.out + KLB_OFF, sRed[0]);
  __syncthreads();
  sRed[tid] = klg; __syncthreads();
  for (int s = 128; s > 0; s >>= 1) { if (tid < s) sRed[tid] += sRed[tid + s]; __syncthreads(); }
  if (tid == 0) atomicAdd(a.out + KLG_OFF, sRed[0]);
}

// ---------------- weight convert: f32 W[K][N] -> bf16 Wt[N][K] (transposed) ----------------
__global__ void convw_kernel(const float* __restrict__ w, bf16_t* __restrict__ wt, int K, int N)
{
  int idx = blockIdx.x * 256 + threadIdx.x;
  if (idx < K * N) {
    int k = idx / N, n = idx - k * N;
    wt[(size_t)n * K + k] = (bf16_t)w[idx];
  }
}

__global__ void zero_kl(float* out) { out[KLB_OFF] = 0.f; out[KLG_OFF] = 0.f; }

// ---------------- GCN (tiny: <1% of FLOPs) ----------------
__global__ void __launch_bounds__(256) gcn_kernel(const float* __restrict__ x,
                                                  const float* __restrict__ w1,
                                                  const float* __restrict__ w2,
                                                  const float* __restrict__ alpha,
                                                  float* __restrict__ Hout)
{
  const int b = blockIdx.x, tid = threadIdx.x;
  __shared__ float A[WSZ * WSZ];
  __shared__ float X[WSZ * 256];
  __shared__ float T[WSZ * 256];
  __shared__ float H1[WSZ * 128];
  for (int i = tid; i < WSZ * WSZ; i += 256) A[i] = ((i / WSZ) == (i % WSZ)) ? 1.f : 0.f;
  __syncthreads();
  for (int p = tid; p < PPAIR; p += 256) {
    int j = 1; while (((j * (j + 1)) >> 1) <= p) ++j;
    int k = p - ((j * (j - 1)) >> 1);
    float v = alpha[(size_t)b * PPAIR + p];
    A[j * WSZ + k] = v; A[k * WSZ + j] = v;
  }
  for (int i = tid; i < WSZ * 256; i += 256) X[i] = x[(size_t)b * WSZ * 256 + i];
  __syncthreads();
  for (int i = tid; i < WSZ * 256; i += 256) {          // T = A @ X
    int n = i >> 8, d = i & 255;
    float s = 0.f;
    for (int m = 0; m < WSZ; ++m) s += A[n * WSZ + m] * X[m * 256 + d];
    T[i] = s;
  }
  __syncthreads();
  for (int i = tid; i < WSZ * 128; i += 256) {          // H1 = relu(T @ W1)
    int n = i >> 7, d = i & 127;
    float s = 0.f;
    for (int k = 0; k < 256; ++k) s += T[n * 256 + k] * w1[k * 128 + d];
    H1[i] = fmaxf(s, 0.f);
  }
  __syncthreads();
  for (int i = tid; i < WSZ * 128; i += 256) {          // X[:,:128] = A @ H1
    int n = i >> 7, d = i & 127;
    float s = 0.f;
    for (int m = 0; m < WSZ; ++m) s += A[n * WSZ + m] * H1[m * 128 + d];
    X[i] = s;
  }
  __syncthreads();
  for (int i = tid; i < WSZ * 128; i += 256) {          // out = relu((A@H1) @ W2)
    int n = i >> 7, d = i & 127;
    float s = 0.f;
    for (int k = 0; k < 128; ++k) s += X[n * 128 + k] * w2[k * 128 + d];
    Hout[(size_t)b * WSZ * 128 + i] = fmaxf(s, 0.f);
  }
}

extern "C" void kernel_launch(void* const* d_in, const int* in_sizes, int n_in,
                              void* d_out, int out_size, void* d_ws, size_t ws_size,
                              hipStream_t stream)
{
  // d_in flattened dict order:
  // 0:x  1:e0.w 2:e0.b 3:e1.w 4:e1.b 5:e2.w 6:e2.b
  // 7:pr.enc.w 8:b 9:pr.enc_g.w 10:b 11:pr.enc_b.w 12:b 13:pr.mean.w 14:b 15:pr.std.w 16:b
  // 17:prior_mij.w 18:b
  // 19:po.enc.w 20:b 21:po.enc_g.w 22:b 23:po.enc_b.w 24:b 25:po.mean.w 26:b 27:po.std.w 28:b
  // 29:pma.w 30:b 31:psa.w 32:b 33:graph_trans.w 34:b 35:gcn1_w 36:gcn2_w 37:eps1
  char* ws = (char*)d_ws;
  const int widx[11] = {1, 3, 5, 7, 9, 11, 19, 21, 23, 29, 31};
  const int bidx[11] = {2, 4, 6, 8, 10, 12, 20, 22, 24, 30, 32};
  const int kdim[11] = {512, 256, 256, 256, 256, 256, 256, 256, 256, 256, 256};
  bf16_t* wt[11];
  size_t off = 0;
  for (int i = 0; i < 11; ++i) {
    wt[i] = (bf16_t*)(ws + off);
    int total = kdim[i] * 256;
    off += (size_t)total * sizeof(bf16_t);
    convw_kernel<<<(total + 255) / 256, 256, 0, stream>>>((const float*)d_in[widx[i]],
                                                          wt[i], kdim[i], 256);
  }
  zero_kl<<<1, 1, 0, stream>>>((float*)d_out);

  MegaArgs a;
  a.x    = (const float*)d_in[0];
  a.eps1 = (const float*)d_in[37];
  a.out  = (float*)d_out;
  for (int i = 0; i < 11; ++i) { a.L[i].wt = wt[i]; a.L[i].b = (const float*)d_in[bidx[i]]; }
  a.pmean.w  = (const float*)d_in[13]; a.pmean.b  = (const float*)d_in[14];
  a.pstd.w   = (const float*)d_in[15]; a.pstd.b   = (const float*)d_in[16];
  a.pmij.w   = (const float*)d_in[17]; a.pmij.b   = (const float*)d_in[18];
  a.qmean.w  = (const float*)d_in[25]; a.qmean.b  = (const float*)d_in[26];
  a.qstd.w   = (const float*)d_in[27]; a.qstd.b   = (const float*)d_in[28];
  a.gtrans.w = (const float*)d_in[33]; a.gtrans.b = (const float*)d_in[34];
  mega_kernel<<<NROWS / 16, 256, 0, stream>>>(a);

  gcn_kernel<<<BATCH, 256, 0, stream>>>((const float*)d_in[0],
                                        (const float*)d_in[35], (const float*)d_in[36],
                                        (const float*)d_out + ALPHA_OFF, (float*)d_out);
}
